// RCGNLayer_41506563948628
// MI455X (gfx1250) — matrix operand order repaired and checked by reading, not compiled, not run on hardware
//
#include <hip/hip_runtime.h>

typedef __attribute__((ext_vector_type(16))) _Float16 v16h;
typedef __attribute__((ext_vector_type(8)))  float    v8f;
typedef __attribute__((ext_vector_type(4)))  float    v4f;

#define N_NODES 50000
#define N_REL   8
#define N_EDGES 800000
#define D       128
#define M_TILES 3125   // 50000 / 16

// ---------------------------------------------------------------------------
// Kernel 1: H[r] = X @ W[r] for r = 0..7 (into workspace),
//           r == 8: out = X @ W0 (self-loop term, initializes d_out).
// One 256-thread block = 8 waves; wave w computes a 16x128 output stripe
// as 8 WMMA 16x16 tiles, K = 128 = 4 chunks of 32 (f16 in, f32 accumulate).
// W tile is staged in LDS pre-converted to f16 in WMMA-B fragment layout.
// ---------------------------------------------------------------------------
__global__ __launch_bounds__(256) void rgcn_gemm(
    const float* __restrict__ X, const float* __restrict__ W,
    const float* __restrict__ W0, float* __restrict__ H,
    float* __restrict__ out)
{
    // Bfrag[kc*8+nt][lane] = 16 halfs of the B fragment for that lane.
    __shared__ v16h Bfrag[32][32];   // 32 KB LDS

    const int r = blockIdx.y;
    const float* __restrict__ Wg = (r < 8) ? (W + (size_t)r * D * D) : W0;
    float* __restrict__ dstbase   = (r < 8) ? (H + (size_t)r * N_NODES * D) : out;

    const int tid = threadIdx.x;

    // Stage weights -> LDS as f16 WMMA-B fragments.
    // ISA 16-bit B layout (32x16): lanes 0-15 hold col n = lane, halfs 0..7 = K 0..7,
    // halfs 8..15 = K 16..23; lanes 16-31 hold col n = lane-16, K 8..15 / 24..31.
    for (int s = tid; s < 1024; s += 256) {
        const int frag = s >> 5;          // kc*8 + nt
        const int lane = s & 31;
        const int kc   = frag >> 3;
        const int nt   = frag & 7;
        const int col  = nt * 16 + (lane & 15);
        const int k0   = kc * 32 + ((lane & 16) ? 8 : 0);
        v16h b;
        #pragma unroll
        for (int h = 0; h < 16; ++h)
            b[h] = (_Float16)Wg[(size_t)(k0 + h + ((h & 8) ? 8 : 0)) * D + col];
        Bfrag[frag][lane] = b;
    }
    __syncthreads();

    const int wave  = tid >> 5;
    const int lane  = tid & 31;
    const int mtile = blockIdx.x * 8 + wave;
    if (mtile >= M_TILES) return;

    // ISA 16-bit A layout (16x32): lanes 0-15: row = lane, halfs 0..7 = K 0..7,
    // halfs 8..15 = K 16..23. Lanes 16-31: row = lane-16, K 8..15 / 24..31.
    const int row  = mtile * 16 + (lane & 15);
    const int ksel = (lane & 16) ? 8 : 0;
    const float* __restrict__ xrow = X + (size_t)row * D;

    v8f acc[8] = {};

    #pragma unroll
    for (int kc = 0; kc < 4; ++kc) {
        const float* g0 = xrow + kc * 32 + ksel;
        const v4f a0 = *(const v4f*)(g0);
        const v4f a1 = *(const v4f*)(g0 + 4);
        const v4f a2 = *(const v4f*)(g0 + 16);
        const v4f a3 = *(const v4f*)(g0 + 20);
        v16h a;
        #pragma unroll
        for (int j = 0; j < 4; ++j) {
            a[j]      = (_Float16)a0[j];
            a[4 + j]  = (_Float16)a1[j];
            a[8 + j]  = (_Float16)a2[j];
            a[12 + j] = (_Float16)a3[j];
        }
        #pragma unroll
        for (int nt = 0; nt < 8; ++nt) {
            const v16h b = Bfrag[kc * 8 + nt][lane];
            acc[nt] = __builtin_amdgcn_wmma_f32_16x16x32_f16(
                /*neg_a=*/false, a, /*neg_b=*/false, b,
                /*c_mod=*/(short)0, acc[nt],
                /*reuse_a=*/false, /*reuse_b=*/false);
        }
    }

    // C/D layout: VGPR v, lanes 0-15 -> M = v, lanes 16-31 -> M = v+8; N = lane%16.
    const int crow0 = mtile * 16 + ((lane & 16) ? 8 : 0);
    #pragma unroll
    for (int nt = 0; nt < 8; ++nt) {
        const int col = nt * 16 + (lane & 15);
        #pragma unroll
        for (int v = 0; v < 8; ++v)
            dstbase[(size_t)(crow0 + v) * D + col] = acc[nt][v];
    }
}

// ---------------------------------------------------------------------------
// Kernel 2: fused gather + scale + segment-sum via hardware fp32 atomics.
// One wave per edge; each lane moves a float4 of the 128-float message.
// ---------------------------------------------------------------------------
__global__ __launch_bounds__(256) void rgcn_edges(
    const float* __restrict__ H, const float* __restrict__ inv_norm,
    const int* __restrict__ srcIdx, const int* __restrict__ dstIdx,
    const int* __restrict__ relIdx, float* __restrict__ out)
{
    const int tid = threadIdx.x;
    const int e   = blockIdx.x * 8 + (tid >> 5);
    if (e >= N_EDGES) return;
    const int lane = tid & 31;

    const int s = srcIdx[e];
    const int d = dstIdx[e];
    const int r = relIdx[e];
    const float coeff = inv_norm[(size_t)d * N_REL + r];

    const v4f* __restrict__ hp =
        (const v4f*)(H + ((size_t)r * N_NODES + s) * D);
    const v4f m = hp[lane] * coeff;

    float* __restrict__ op = out + (size_t)d * D + lane * 4;
    #pragma unroll
    for (int j = 0; j < 4; ++j)
        __hip_atomic_fetch_add(op + j, m[j], __ATOMIC_RELAXED,
                               __HIP_MEMORY_SCOPE_AGENT);
}

// ---------------------------------------------------------------------------
extern "C" void kernel_launch(void* const* d_in, const int* in_sizes, int n_in,
                              void* d_out, int out_size, void* d_ws, size_t ws_size,
                              hipStream_t stream) {
    (void)in_sizes; (void)n_in; (void)out_size; (void)ws_size;

    const float* X        = (const float*)d_in[0];   // (1, 50000, 128)
    const float* W        = (const float*)d_in[1];   // (8, 128, 128)
    const float* W0       = (const float*)d_in[2];   // (128, 128)
    const float* inv_norm = (const float*)d_in[3];   // (50000, 8)
    const int*   srcIdx   = (const int*)d_in[4];     // (800000,)
    const int*   dstIdx   = (const int*)d_in[5];
    const int*   relIdx   = (const int*)d_in[6];

    float* out = (float*)d_out;                      // (1, 50000, 128)
    float* H   = (float*)d_ws;                       // 8*50000*128 fp32 = 204.8 MB

    // Pass 1: 9 GEMMs (8 relations -> ws, self-loop -> out). Fully overwrites
    // d_out each call, so replays are self-initializing.
    dim3 g1((M_TILES + 7) / 8, N_REL + 1);
    rgcn_gemm<<<g1, 256, 0, stream>>>(X, W, W0, H, out);

    // Pass 2: edge gather + scatter-add.
    rgcn_edges<<<(N_EDGES + 7) / 8, 256, 0, stream>>>(H, inv_norm, srcIdx,
                                                      dstIdx, relIdx, out);
}
